// CrossAttentionGating_28595892257252
// MI455X (gfx1250) — matrix operand order repaired and checked by reading, not compiled, not run on hardware
//
#include <hip/hip_runtime.h>

// CrossAttentionGating for MI455X (gfx1250, wave32, WMMA).
// B=8, N=1024, Dv=768, Ds=512, F=768, H=12, dh=64.
// Single-query attention collapse: only the visual-gate GEMM (8192x768x768)
// is large; it runs on v_wmma_f32_16x16x32_bf16 with a fused sigmoid epilogue.
// Gate GEMM uses 32x64 wave tiles with register double-buffering so fragment
// loads for K-step k+32 overlap the WMMAs of K-step k.

#define B_   8
#define N_   1024
#define DV   768
#define DS   512
#define FF   768
#define H_   12
#define DH   64
#define ROWS (B_ * N_)   // 8192

typedef __attribute__((ext_vector_type(16))) __bf16         v16bf;
typedef __attribute__((ext_vector_type(8)))  float          v8f;
typedef __attribute__((ext_vector_type(4)))  unsigned int   u32x4;
typedef __attribute__((ext_vector_type(8)))  unsigned short ushort8;

__device__ __forceinline__ unsigned short f2bf(float f) {
    unsigned int u = __float_as_uint(f);
    unsigned int r = u + 0x7FFFu + ((u >> 16) & 1u);   // round-to-nearest-even
    return (unsigned short)(r >> 16);
}
__device__ __forceinline__ float bf2f(unsigned short h) {
    return __uint_as_float(((unsigned int)h) << 16);
}

// ---------------------------------------------------------------------------
// K1: LayerNorm visual rows -> bf16; also raw visual -> bf16 (gate GEMM A).
// One 256-thread block per row of 768.
// ---------------------------------------------------------------------------
__global__ void __launch_bounds__(256)
ln_prep_kernel(const float* __restrict__ vis,
               const float* __restrict__ gv, const float* __restrict__ bv,
               unsigned short* __restrict__ visn,
               unsigned short* __restrict__ visraw) {
    __shared__ float red[256];
    __shared__ float red2[256];
    const int row = blockIdx.x;
    const int tid = threadIdx.x;
    const float* x = vis + (size_t)row * DV;

    float vals[3];
    float s = 0.f, s2 = 0.f;
#pragma unroll
    for (int i = 0; i < 3; ++i) {
        int c = tid + i * 256;
        float v = x[c];
        vals[i] = v; s += v; s2 += v * v;
    }
    red[tid] = s; red2[tid] = s2;
    __syncthreads();
    for (int off = 128; off > 0; off >>= 1) {
        if (tid < off) { red[tid] += red[tid + off]; red2[tid] += red2[tid + off]; }
        __syncthreads();
    }
    float mu   = red[0] * (1.f / DV);
    float var  = red2[0] * (1.f / DV) - mu * mu;
    float rstd = rsqrtf(var + 1e-5f);
#pragma unroll
    for (int i = 0; i < 3; ++i) {
        int c = tid + i * 256;
        float xn = (vals[i] - mu) * rstd * gv[c] + bv[c];
        visn[(size_t)row * DV + c]   = f2bf(xn);
        visraw[(size_t)row * DV + c] = f2bf(vals[i]);
    }
}

// ---------------------------------------------------------------------------
// K2: convert Wg[:, 0:768] -> compact bf16 (768x768) for WMMA B operand.
// ---------------------------------------------------------------------------
__global__ void __launch_bounds__(256)
wg_conv_kernel(const float* __restrict__ Wg, unsigned short* __restrict__ wgv) {
    int idx = blockIdx.x * 256 + threadIdx.x;      // FF*DV total
    int f = idx / DV, dv = idx - f * DV;
    wgv[idx] = f2bf(Wg[(size_t)f * (DV + DS) + dv]);
}

// ---------------------------------------------------------------------------
// K3: per-batch small math: LN(semantic), q = sem_n @ Wq^T,
//     wqk[h,dv] = sum_d q[h,d]*Wk[h*64+d, dv],
//     semg[f]   = bg[f] + raw_sem @ Wg[:, 768:1280]^T   (gate bias per batch)
// One 256-thread block per batch.
// ---------------------------------------------------------------------------
__global__ void __launch_bounds__(256)
sem_kernel(const float* __restrict__ sem,
           const float* __restrict__ Wq, const float* __restrict__ Wk,
           const float* __restrict__ Wg, const float* __restrict__ bg,
           const float* __restrict__ gs, const float* __restrict__ bs,
           float* __restrict__ wqk, float* __restrict__ semg) {
    __shared__ float semn[DS];
    __shared__ float qs[FF];
    __shared__ float red[256];
    __shared__ float red2[256];
    const int b = blockIdx.x, tid = threadIdx.x;
    const float* x = sem + (size_t)b * DS;

    float vals[2];
    float s = 0.f, s2 = 0.f;
#pragma unroll
    for (int i = 0; i < 2; ++i) {
        int c = tid + i * 256;
        float v = x[c];
        vals[i] = v; s += v; s2 += v * v;
    }
    red[tid] = s; red2[tid] = s2;
    __syncthreads();
    for (int off = 128; off > 0; off >>= 1) {
        if (tid < off) { red[tid] += red[tid + off]; red2[tid] += red2[tid + off]; }
        __syncthreads();
    }
    float mu   = red[0] * (1.f / DS);
    float var  = red2[0] * (1.f / DS) - mu * mu;
    float rstd = rsqrtf(var + 1e-5f);
#pragma unroll
    for (int i = 0; i < 2; ++i) {
        int c = tid + i * 256;
        semn[c] = (vals[i] - mu) * rstd * gs[c] + bs[c];
    }
    __syncthreads();

    // q[f] = semn . Wq[f,:]
    for (int f = tid; f < FF; f += 256) {
        const float* wr = Wq + (size_t)f * DS;
        float acc = 0.f;
        for (int d = 0; d < DS; ++d) acc += semn[d] * wr[d];
        qs[f] = acc;
    }
    __syncthreads();

    // wqk[h,dv] = sum_d qs[h*64+d] * Wk[(h*64+d), dv]
    for (int o = tid; o < H_ * DV; o += 256) {
        int h = o / DV, dv = o - h * DV;
        float acc = 0.f;
        for (int d = 0; d < DH; ++d)
            acc += qs[h * DH + d] * Wk[(size_t)(h * DH + d) * DV + dv];
        wqk[(size_t)b * H_ * DV + o] = acc;
    }

    // semg[f] = bg[f] + raw_sem . Wg[f, 768:1280]
    for (int f = tid; f < FF; f += 256) {
        const float* wr = Wg + (size_t)f * (DV + DS) + DV;
        float acc = bg[f];
        for (int d = 0; d < DS; ++d) acc += x[d] * wr[d];
        semg[(size_t)b * FF + f] = acc;
    }
}

// ---------------------------------------------------------------------------
// K4: per-batch attention: scores = vis_n . wqk, softmax over N,
//     vbar[h,dv] = sum_n attn[h,n] * vis_n[n,dv].
// One 1024-thread block per batch; heads processed 4 at a time (LDS < 64KB).
// ---------------------------------------------------------------------------
__global__ void __launch_bounds__(1024)
attn_kernel(const unsigned short* __restrict__ visn,
            const float* __restrict__ wqk,
            float* __restrict__ vbar) {
    __shared__ float wqks[H_ * DV];   // 36 KB
    __shared__ float attn[4 * N_];    // 16 KB
    __shared__ float red[N_];         //  4 KB
    const int b = blockIdx.x, tid = threadIdx.x;

    for (int i = tid; i < H_ * DV; i += 1024) wqks[i] = wqk[(size_t)b * H_ * DV + i];
    __syncthreads();

    const ushort8* xr8 = (const ushort8*)(visn + ((size_t)b * N_ + tid) * DV);

    for (int g = 0; g < 3; ++g) {          // head groups of 4
        // ---- scores for heads 4g..4g+3 (thread = row n) ----
        float acc[4] = {0.f, 0.f, 0.f, 0.f};
        for (int c = 0; c < DV / 8; ++c) {
            ushort8 pk = xr8[c];
#pragma unroll
            for (int i = 0; i < 8; ++i) {
                float xv = bf2f(pk[i]);
                int dv = c * 8 + i;
#pragma unroll
                for (int hh = 0; hh < 4; ++hh)
                    acc[hh] += xv * wqks[(g * 4 + hh) * DV + dv];
            }
        }
#pragma unroll
        for (int hh = 0; hh < 4; ++hh) attn[hh * N_ + tid] = acc[hh] * 0.125f; // dh^-0.5
        __syncthreads();

        // ---- softmax over n for each of the 4 heads ----
        for (int hh = 0; hh < 4; ++hh) {
            float v = attn[hh * N_ + tid];
            red[tid] = v; __syncthreads();
            for (int off = 512; off > 0; off >>= 1) {
                if (tid < off) red[tid] = fmaxf(red[tid], red[tid + off]);
                __syncthreads();
            }
            float mx = red[0]; __syncthreads();
            float e = __expf(v - mx);
            red[tid] = e; __syncthreads();
            for (int off = 512; off > 0; off >>= 1) {
                if (tid < off) red[tid] += red[tid + off];
                __syncthreads();
            }
            float sm = red[0]; __syncthreads();
            attn[hh * N_ + tid] = e / sm;
        }
        __syncthreads();

        // ---- vbar for these heads ----
        for (int o = tid; o < 4 * DV; o += 1024) {
            int hh = o / DV, dv = o - hh * DV;
            float a = 0.f;
            for (int n = 0; n < N_; ++n)
                a += attn[hh * N_ + n] * bf2f(visn[((size_t)b * N_ + n) * DV + dv]);
            vbar[(size_t)b * H_ * DV + (g * 4 + hh) * DV + dv] = a;
        }
        __syncthreads();
    }
}

// ---------------------------------------------------------------------------
// K5: per-batch: ctx = vbar . Wv rows, attended = ctx @ Wo^T + bo.
// ---------------------------------------------------------------------------
__global__ void __launch_bounds__(256)
attended_kernel(const float* __restrict__ vbar,
                const float* __restrict__ Wv, const float* __restrict__ Wo,
                const float* __restrict__ bo,
                float* __restrict__ attended) {
    __shared__ float ctx[FF];
    __shared__ float vb[H_ * DV];   // 36 KB
    const int b = blockIdx.x, tid = threadIdx.x;
    for (int i = tid; i < H_ * DV; i += 256) vb[i] = vbar[(size_t)b * H_ * DV + i];
    __syncthreads();
    for (int f = tid; f < FF; f += 256) {
        int h = f / DH;
        const float* wr = Wv + (size_t)f * DV;
        float acc = 0.f;
        for (int dv = 0; dv < DV; ++dv) acc += vb[h * DV + dv] * wr[dv];
        ctx[f] = acc;
    }
    __syncthreads();
    for (int j = tid; j < DV; j += 256) {
        const float* wr = Wo + (size_t)j * FF;
        float acc = bo[j];
        for (int f = 0; f < FF; ++f) acc += ctx[f] * wr[f];
        attended[(size_t)b * DV + j] = acc;
    }
}

// ---------------------------------------------------------------------------
// K6: gate GEMM on WMMA + fused sigmoid-gate epilogue.
// out[n,f] = vis[n,f] + sigmoid( visraw[n,:]@Wg_v[f,:] + semg[b,f] ) * attended[b,f]
// Wave tile: 32(M) x 64(N): 2 A-frags + 4 B-frags -> 8 WMMAs per K-step.
// Register double-buffering: K-step k+32 fragments are fetched before the
// WMMAs of K-step k execute, so global-load latency overlaps the matrix pipe.
// Block = 256 threads = 8 waves -> 256 rows x 64 cols per block.
// ---------------------------------------------------------------------------
union Frag { u32x4 u[2]; v16bf v; };

__device__ __forceinline__ void load_a_frag(Frag& f, const unsigned short* arow, int k0, bool hi) {
    int ak = k0 + (hi ? 8 : 0);
    f.u[0] = *(const u32x4*)(arow + ak);        // K = ak .. ak+7
    f.u[1] = *(const u32x4*)(arow + ak + 16);   // K = ak+16 .. ak+23
}
__device__ __forceinline__ void load_b_frag(Frag& f, const unsigned short* brow, int k0, bool hi) {
    int bk = k0 + (hi ? 8 : 0);
    f.u[0] = *(const u32x4*)(brow + bk);        // K = bk .. bk+7
    f.u[1] = *(const u32x4*)(brow + bk + 16);   // K = bk+16 .. bk+23
}

__global__ void __launch_bounds__(256)
gate_wmma_kernel(const unsigned short* __restrict__ visraw,  // ROWS x DV bf16
                 const unsigned short* __restrict__ wgv,     // FF x DV bf16
                 const float* __restrict__ semg,             // B x FF
                 const float* __restrict__ attended,         // B x DV
                 const float* __restrict__ vis,              // ROWS x DV f32
                 float* __restrict__ out) {
    const int wave = threadIdx.x >> 5;
    const int lane = threadIdx.x & 31;
    const int m0 = blockIdx.x * 256 + wave * 32;   // 32 rows per wave
    const int n0 = blockIdx.y * 64;
    const int b  = m0 >> 10;                       // 1024 rows per batch
    const bool hi = lane >= 16;
    const int l15 = lane & 15;

    v8f acc[2][4];
#pragma unroll
    for (int mi = 0; mi < 2; ++mi)
#pragma unroll
        for (int sub = 0; sub < 4; ++sub) {
            v8f z = {0.f, 0.f, 0.f, 0.f, 0.f, 0.f, 0.f, 0.f};
            acc[mi][sub] = z;
        }

    const unsigned short* arow[2] = {
        visraw + (size_t)(m0 + l15) * DV,
        visraw + (size_t)(m0 + 16 + l15) * DV
    };
    const unsigned short* brow[4] = {
        wgv + (size_t)(n0 +  0 + l15) * DV,
        wgv + (size_t)(n0 + 16 + l15) * DV,
        wgv + (size_t)(n0 + 32 + l15) * DV,
        wgv + (size_t)(n0 + 48 + l15) * DV
    };

    Frag a_cur[2], b_cur[4], a_nxt[2], b_nxt[4];
#pragma unroll
    for (int mi = 0; mi < 2; ++mi) load_a_frag(a_cur[mi], arow[mi], 0, hi);
#pragma unroll
    for (int sub = 0; sub < 4; ++sub) load_b_frag(b_cur[sub], brow[sub], 0, hi);

#pragma unroll
    for (int k0 = 0; k0 < DV; k0 += 32) {
        const int kn = k0 + 32;
        if (kn < DV) {
            // Prefetch next K-step fragments before consuming current ones.
#pragma unroll
            for (int mi = 0; mi < 2; ++mi) load_a_frag(a_nxt[mi], arow[mi], kn, hi);
#pragma unroll
            for (int sub = 0; sub < 4; ++sub) load_b_frag(b_nxt[sub], brow[sub], kn, hi);
        }
#pragma unroll
        for (int mi = 0; mi < 2; ++mi)
#pragma unroll
            for (int sub = 0; sub < 4; ++sub)
                acc[mi][sub] = __builtin_amdgcn_wmma_f32_16x16x32_bf16(
                    false, a_cur[mi].v, false, b_cur[sub].v, (short)0,
                    acc[mi][sub], false, false);
        if (kn < DV) {
#pragma unroll
            for (int mi = 0; mi < 2; ++mi) a_cur[mi] = a_nxt[mi];
#pragma unroll
            for (int sub = 0; sub < 4; ++sub) b_cur[sub] = b_nxt[sub];
        }
    }

    // Epilogue per C/D layout: lane<16 -> N=lane, M=r; lane>=16 -> M=8+r.
#pragma unroll
    for (int sub = 0; sub < 4; ++sub) {
        int col = n0 + sub * 16 + l15;
        float sg = semg[(size_t)b * FF + col];
        float at = attended[(size_t)b * DV + col];
#pragma unroll
        for (int mi = 0; mi < 2; ++mi) {
#pragma unroll
            for (int r = 0; r < 8; ++r) {
                int row = m0 + mi * 16 + (hi ? 8 + r : r);
                float pre = acc[mi][sub][r] + sg;
                float gte = 1.f / (1.f + __expf(-pre));
                size_t idx = (size_t)row * DV + col;
                out[idx] = vis[idx] + gte * at;
            }
        }
    }
}

// ---------------------------------------------------------------------------
// Launch
// ---------------------------------------------------------------------------
extern "C" void kernel_launch(void* const* d_in, const int* in_sizes, int n_in,
                              void* d_out, int out_size, void* d_ws, size_t ws_size,
                              hipStream_t stream) {
    const float* vis = (const float*)d_in[0];
    const float* sem = (const float*)d_in[1];
    const float* Wq  = (const float*)d_in[2];
    const float* Wk  = (const float*)d_in[3];
    const float* Wv  = (const float*)d_in[4];
    const float* Wo  = (const float*)d_in[5];
    const float* bo  = (const float*)d_in[6];
    const float* Wg  = (const float*)d_in[7];
    const float* bg  = (const float*)d_in[8];
    const float* gv  = (const float*)d_in[9];
    const float* bv  = (const float*)d_in[10];
    const float* gs  = (const float*)d_in[11];
    const float* bs  = (const float*)d_in[12];
    float* out = (float*)d_out;

    // Workspace layout (16B-aligned offsets), ~27 MB total.
    char* ws = (char*)d_ws;
    unsigned short* visn     = (unsigned short*)(ws);                 // 12,582,912 B
    unsigned short* visraw   = (unsigned short*)(ws + 12582912);      // 12,582,912 B
    unsigned short* wgv      = (unsigned short*)(ws + 25165824);      //  1,179,648 B
    float*          wqk      = (float*)(ws + 26345472);               //    294,912 B
    float*          semg     = (float*)(ws + 26640384);               //     24,576 B
    float*          vbar     = (float*)(ws + 26664960);               //    294,912 B
    float*          attended = (float*)(ws + 26959872);               //     24,576 B

    ln_prep_kernel<<<dim3(ROWS), dim3(256), 0, stream>>>(vis, gv, bv, visn, visraw);
    wg_conv_kernel<<<dim3((FF * DV) / 256), dim3(256), 0, stream>>>(Wg, wgv);
    sem_kernel<<<dim3(B_), dim3(256), 0, stream>>>(sem, Wq, Wk, Wg, bg, gs, bs, wqk, semg);
    attn_kernel<<<dim3(B_), dim3(1024), 0, stream>>>(visn, wqk, vbar);
    attended_kernel<<<dim3(B_), dim3(256), 0, stream>>>(vbar, Wv, Wo, bo, attended);
    gate_wmma_kernel<<<dim3(ROWS / 256, FF / 64), dim3(256), 0, stream>>>(
        visraw, wgv, semg, attended, vis, out);
}
